// FullEnc_11278584119826
// MI455X (gfx1250) — matrix operand order repaired and checked by reading, not compiled, not run on hardware
//
#include <hip/hip_runtime.h>
#include <hip/hip_bf16.h>
#include <hip/hip_fp16.h>

typedef __attribute__((ext_vector_type(8)))  _Float16 v8h;
typedef __attribute__((ext_vector_type(16))) _Float16 v16h;
typedef __attribute__((ext_vector_type(8)))  float    v8f;

#define NN 65536
#define HH 300     // hidden
#define HP 320     // hidden padded (10 k-tiles of 32)
#define DD 100     // d_model
#define DP 112     // d_model padded (7 n-tiles of 16)
#define FEAT_W 704 // 7*100 padded to 22 k-tiles of 32

// ---------------- f16 weight region layout (element offsets in ws) ----------
static const size_t OFF_BOXW1 = 0;                         // 64x320
static const size_t OFF_BOXW2 = OFF_BOXW1 + 64*320;        // 320x112
static const size_t OFF_OFFW1 = OFF_BOXW2 + 320*112;       // 32x320
static const size_t OFF_OFFW2 = OFF_OFFW1 + 32*320;        // 320x64
static const size_t OFF_KOW1  = OFF_OFFW2 + 320*64;        // 192x320
static const size_t OFF_KOW2  = OFF_KOW1  + 192*320;       // 320x16
static const size_t OFF_WM1   = OFF_KOW2  + 320*16;        // 6 x 224x320
static const size_t OFF_WM2   = OFF_WM1   + 6*224*320;     // 6 x 320x112
static const size_t OFF_WH1   = OFF_WM2   + 6*320*112;     // 6 x 128x320
static const size_t OFF_WH2   = OFF_WH1   + 6*128*320;     // 6 x 320x112
static const size_t OFF_WX1   = OFF_WH2   + 6*320*112;     // 6 x 128x320
static const size_t OFF_WX2   = OFF_WX1   + 6*128*320;     // 6 x 320x112
static const size_t OFF_UPW1  = OFF_WX2   + 6*320*112;     // 704x320
static const size_t OFF_UPW2  = OFF_UPW1  + 704*320;       // 320x112
static const size_t OFF_WEND  = OFF_UPW2  + 320*112;       // 1,981,440 f16
static const size_t OFF_OFFBUF = (OFF_WEND + 127) & ~(size_t)127;   // N x 64 f16
static const size_t OFF_FEAT   = OFF_OFFBUF + (size_t)NN*64;        // N x 704 f16
static const size_t OFF_F16END = OFF_FEAT + (size_t)NN*FEAT_W;

// ---------------- weight convert: fp32 -> f16, zero-padded ------------------
__global__ void k_cvt(const float* __restrict__ src, _Float16* __restrict__ dst,
                      int Kdim, int Ncols, int Kpad, int Npad, int nmat) {
    int total = nmat * Kpad * Npad;
    for (int idx = blockIdx.x * blockDim.x + threadIdx.x; idx < total;
         idx += gridDim.x * blockDim.x) {
        int ms = Kpad * Npad;
        int m  = idx / ms;
        int r  = idx - m * ms;
        int k  = r / Npad;
        int n  = r - k * Npad;
        float v = (k < Kdim && n < Ncols) ? src[((size_t)m * Kdim + k) * Ncols + n] : 0.0f;
        dst[idx] = (_Float16)v;
    }
}

// ---------------- WMMA fragment loaders -------------------------------------
// A (16x32 f16): lane holds row (lane&15); K elems {base..base+7, 16+base..16+base+7},
// base = (lane>>4)*8  -- matches ISA 16-bit A layout.
__device__ __forceinline__ v16h load_a_tile(const _Float16* base, int stride, int kt, int lane) {
    int arow  = lane & 15;
    int abase = (lane >> 4) * 8;
    const _Float16* p = base + arow * stride + kt * 32 + abase;
    v8h lo = *(const v8h*)p;
    v8h hi = *(const v8h*)(p + 16);
    return __builtin_shufflevector(lo, hi, 0,1,2,3,4,5,6,7,8,9,10,11,12,13,14,15);
}

// B (32x16 f16): lane holds col nt*16+(lane&15); K elems {base..base+7, 16+base..16+base+7}, base = (lane>>4)*8.
__device__ __forceinline__ v16h load_b_g(const _Float16* __restrict__ W, int Npad,
                                         int kt, int nt, int lane) {
    v16h b;
    const _Float16* p = W + (size_t)(kt * 32 + (lane >> 4) * 8) * Npad + nt * 16 + (lane & 15);
#pragma unroll
    for (int s = 0; s < 16; ++s)
        b[s] = p[(size_t)((s >> 3) * 16 + (s & 7)) * Npad];
    return b;
}

// ---------------- fused Linear+bias+ReLU -> LDS hidden (f16, width HP) ------
__device__ __forceinline__ void mlp_hidden(const _Float16* __restrict__ W1,
                                           const float* __restrict__ b1,
                                           const _Float16* Xl, int KPAD,
                                           _Float16* Hl, int lane) {
    int half = lane >> 4, colin = lane & 15;
    for (int nt = 0; nt < HP / 16; ++nt) {
        v8f c = {};
        for (int kt = 0; kt < KPAD / 32; ++kt) {
            v16h a = load_a_tile(Xl, KPAD, kt, lane);
            v16h b = load_b_g(W1, HP, kt, nt, lane);
            c = __builtin_amdgcn_wmma_f32_16x16x32_f16(false, a, false, b,
                                                       (short)0, c, false, false);
        }
        int col = nt * 16 + colin;
        float bias = (col < HH) ? b1[col] : 0.0f;
#pragma unroll
        for (int i = 0; i < 8; ++i) {
            float v = c[i] + bias;
            Hl[(i + 8 * half) * HP + col] = (_Float16)(v > 0.0f ? v : 0.0f);
        }
    }
}

// ---------------- second layer accumulator for one 16-col tile --------------
__device__ __forceinline__ v8f mlp_out_tile(const _Float16* __restrict__ W2, int Npad,
                                            const _Float16* Hl, int nt, int lane) {
    v8f c = {};
    for (int kt = 0; kt < HP / 32; ++kt) {
        v16h a = load_a_tile(Hl, HP, kt, lane);
        v16h b = load_b_g(W2, Npad, kt, nt, lane);
        c = __builtin_amdgcn_wmma_f32_16x16x32_f16(false, a, false, b,
                                                   (short)0, c, false, false);
    }
    return c;
}

// ---------------- X loader: fp32 global -> f16 LDS, zero-padded -------------
__device__ __forceinline__ void load_x_f32(_Float16* Xl, const float* __restrict__ src,
                                           int row0, int srcStride, int Kdim, int KPAD,
                                           int lane) {
    for (int i = lane; i < 16 * KPAD; i += 32) {
        int r = i / KPAD, c = i - r * KPAD;
        float v = (c < Kdim) ? src[(size_t)(row0 + r) * srcStride + c] : 0.0f;
        Xl[i] = (_Float16)v;
    }
}

// ---------------- BoxEnc: self_vec = mlp(k_vec) -> feat[:,0:100] ------------
__global__ void __launch_bounds__(64)
k_box(const float* __restrict__ kvec, const _Float16* __restrict__ W1,
      const float* __restrict__ b1, const _Float16* __restrict__ W2,
      const float* __restrict__ b2, _Float16* __restrict__ feat) {
    __shared__ alignas(32) _Float16 X[2][16 * 64];
    __shared__ alignas(32) _Float16 Hid[2][16 * HP];
    int wave = threadIdx.x >> 5, lane = threadIdx.x & 31;
    int row0 = (blockIdx.x * 2 + wave) * 16;
    load_x_f32(X[wave], kvec, row0, 55, 55, 64, lane);
    mlp_hidden(W1, b1, X[wave], 64, Hid[wave], lane);
    int half = lane >> 4, colin = lane & 15;
    for (int nt = 0; nt < DP / 16; ++nt) {
        v8f c = mlp_out_tile(W2, DP, Hid[wave], nt, lane);
        int col = nt * 16 + colin;
        if (col < DD) {
            float bias = b2[col];
#pragma unroll
            for (int i = 0; i < 8; ++i)
                feat[(size_t)(row0 + i + 8 * half) * FEAT_W + col] = (_Float16)(c[i] + bias);
        }
    }
}

// ---------------- off = mlp(offset_vec) -> offbuf [N,64] f16 ----------------
__global__ void __launch_bounds__(64)
k_off(const float* __restrict__ offv, const _Float16* __restrict__ W1,
      const float* __restrict__ b1, const _Float16* __restrict__ W2,
      const float* __restrict__ b2, _Float16* __restrict__ offbuf) {
    __shared__ alignas(32) _Float16 X[2][16 * 32];
    __shared__ alignas(32) _Float16 Hid[2][16 * HP];
    int wave = threadIdx.x >> 5, lane = threadIdx.x & 31;
    int row0 = (blockIdx.x * 2 + wave) * 16;
    load_x_f32(X[wave], offv, row0, 3, 3, 32, lane);
    mlp_hidden(W1, b1, X[wave], 32, Hid[wave], lane);
    int half = lane >> 4, colin = lane & 15;
    for (int nt = 0; nt < 4; ++nt) {                  // Npad = 64
        v8f c = mlp_out_tile(W2, 64, Hid[wave], nt, lane);
        int col = nt * 16 + colin;
        float bias = (col < 55) ? b2[col] : 0.0f;     // pads store exact 0
#pragma unroll
        for (int i = 0; i < 8; ++i)
            offbuf[(size_t)(row0 + i + 8 * half) * 64 + col] = (_Float16)(c[i] + bias);
    }
}

// ---------------- w = sigmoid(mlp(cat(k1,k2,off))) -> wbuf [N] f32 ----------
__global__ void __launch_bounds__(64)
k_ko(const float* __restrict__ kv1, const float* __restrict__ kv2,
     const _Float16* __restrict__ offbuf, const _Float16* __restrict__ W1,
     const float* __restrict__ b1, const _Float16* __restrict__ W2,
     const float* __restrict__ b2, float* __restrict__ wbuf) {
    __shared__ alignas(32) _Float16 X[2][16 * 192];
    __shared__ alignas(32) _Float16 Hid[2][16 * HP];
    int wave = threadIdx.x >> 5, lane = threadIdx.x & 31;
    int row0 = (blockIdx.x * 2 + wave) * 16;
    for (int i = lane; i < 16 * 192; i += 32) {
        int r = i / 192, c = i - r * 192;
        float v;
        if (c < 55)        v = kv1[(size_t)(row0 + r) * 55 + c];
        else if (c < 110)  v = kv2[(size_t)(row0 + r) * 55 + (c - 55)];
        else if (c < 165)  v = (float)offbuf[(size_t)(row0 + r) * 64 + (c - 110)];
        else               v = 0.0f;
        X[wave][i] = (_Float16)v;
    }
    mlp_hidden(W1, b1, X[wave], 192, Hid[wave], lane);
    int half = lane >> 4, colin = lane & 15;
    v8f c = mlp_out_tile(W2, 16, Hid[wave], 0, lane); // Npad = 16, Cout = 1
    if (colin == 0) {
        float bias = b2[0];
#pragma unroll
        for (int i = 0; i < 8; ++i) {
            float v = c[i] + bias;
            wbuf[row0 + i + 8 * half] = 1.0f / (1.0f + __expf(-v));
        }
    }
}

// ---------------- per-relation: msg MLP + two GRU MLPs + w-combine ----------
__global__ void __launch_bounds__(64)
k_rel(const float* __restrict__ cur, const float* __restrict__ dvecs,
      const float* __restrict__ pre, const float* __restrict__ wbuf,
      const _Float16* __restrict__ WM1, const float* __restrict__ bm1,
      const _Float16* __restrict__ WM2, const float* __restrict__ bm2,
      const _Float16* __restrict__ WH1, const float* __restrict__ bh1,
      const _Float16* __restrict__ WH2, const float* __restrict__ bh2,
      const _Float16* __restrict__ WX1, const float* __restrict__ bx1,
      const _Float16* __restrict__ WX2, const float* __restrict__ bx2,
      _Float16* __restrict__ feat) {
    __shared__ alignas(32) _Float16 X[2][16 * 224];
    __shared__ alignas(32) _Float16 HA[2][16 * HP];
    __shared__ alignas(32) _Float16 HB[2][16 * HP];
    __shared__ alignas(32) _Float16 MSG[2][16 * 128];
    int wave = threadIdx.x >> 5, lane = threadIdx.x & 31;
    int r = blockIdx.y;
    int row0 = (blockIdx.x * 2 + wave) * 16;
    const _Float16* wm1 = WM1 + (size_t)r * 224 * HP;
    const _Float16* wm2 = WM2 + (size_t)r * HP * DP;
    const _Float16* wh1 = WH1 + (size_t)r * 128 * HP;
    const _Float16* wh2 = WH2 + (size_t)r * HP * DP;
    const _Float16* wx1 = WX1 + (size_t)r * 128 * HP;
    const _Float16* wx2 = WX2 + (size_t)r * HP * DP;
    const float* bm1r = bm1 + r * HH; const float* bm2r = bm2 + r * DD;
    const float* bh1r = bh1 + r * HH; const float* bh2r = bh2 + r * DD;
    const float* bx1r = bx1 + r * HH; const float* bx2r = bx2 + r * DD;
    const float* dv = dvecs + (size_t)r * NN * DD;

    // X = cat(cur_d_vec, d_vecs[r]), Kpad=224
    for (int i = lane; i < 16 * 224; i += 32) {
        int rr = i / 224, c = i - rr * 224;
        float v = 0.0f;
        if (c < DD)          v = cur[(size_t)(row0 + rr) * DD + c];
        else if (c < 2 * DD) v = dv[(size_t)(row0 + rr) * DD + (c - DD)];
        X[wave][i] = (_Float16)v;
    }
    mlp_hidden(wm1, bm1r, X[wave], 224, HA[wave], lane);

    // msg = hidden @ Wm2 + bm2 -> LDS (f16, width 128, zero-padded)
    for (int i = lane; i < 16 * 128; i += 32) MSG[wave][i] = (_Float16)0.0f;
    int half = lane >> 4, colin = lane & 15;
    for (int nt = 0; nt < DP / 16; ++nt) {
        v8f c = mlp_out_tile(wm2, DP, HA[wave], nt, lane);
        int col = nt * 16 + colin;
        float bias = (col < DD) ? bm2r[col] : 0.0f;
#pragma unroll
        for (int i = 0; i < 8; ++i)
            MSG[wave][(i + 8 * half) * 128 + col] = (_Float16)(c[i] + bias);
    }

    // ht = mlp(pre) + mlp(msg) * w
    load_x_f32(X[wave], pre + (size_t)r * NN * DD, row0, DD, DD, 128, lane);
    mlp_hidden(wh1, bh1r, X[wave], 128, HA[wave], lane);
    mlp_hidden(wx1, bx1r, MSG[wave], 128, HB[wave], lane);
    for (int nt = 0; nt < DP / 16; ++nt) {
        v8f ca = mlp_out_tile(wh2, DP, HA[wave], nt, lane);
        v8f cb = mlp_out_tile(wx2, DP, HB[wave], nt, lane);
        int col = nt * 16 + colin;
        if (col < DD) {
            float ba = bh2r[col], bb = bx2r[col];
            int cb0 = DD + r * DD;
#pragma unroll
            for (int i = 0; i < 8; ++i) {
                int row = row0 + i + 8 * half;
                float wv = wbuf[row];
                feat[(size_t)row * FEAT_W + cb0 + col] =
                    (_Float16)((ca[i] + ba) + wv * (cb[i] + bb));
            }
        }
    }
}

// ---------------- UpdateEnc: out = mlp(feat) -> d_out [N,100] f32 -----------
__global__ void __launch_bounds__(32)
k_up(const _Float16* __restrict__ feat, const _Float16* __restrict__ W1,
     const float* __restrict__ b1, const _Float16* __restrict__ W2,
     const float* __restrict__ b2, float* __restrict__ out) {
    __shared__ alignas(32) _Float16 X[16 * FEAT_W];
    __shared__ alignas(32) _Float16 Hid[16 * HP];
    int lane = threadIdx.x & 31;
    int row0 = blockIdx.x * 16;
    for (int i = lane; i < 16 * FEAT_W; i += 32) {
        int r = i / FEAT_W, c = i - r * FEAT_W;
        X[i] = (c < 700) ? feat[(size_t)(row0 + r) * FEAT_W + c] : (_Float16)0.0f;
    }
    mlp_hidden(W1, b1, X, FEAT_W, Hid, lane);
    int half = lane >> 4, colin = lane & 15;
    for (int nt = 0; nt < DP / 16; ++nt) {
        v8f c = mlp_out_tile(W2, DP, Hid, nt, lane);
        int col = nt * 16 + colin;
        if (col < DD) {
            float bias = b2[col];
#pragma unroll
            for (int i = 0; i < 8; ++i)
                out[(size_t)(row0 + i + 8 * half) * DD + col] = c[i] + bias;
        }
    }
}

// ---------------- host-side orchestration -----------------------------------
extern "C" void kernel_launch(void* const* d_in, const int* in_sizes, int n_in,
                              void* d_out, int out_size, void* d_ws, size_t ws_size,
                              hipStream_t stream) {
    (void)in_sizes; (void)n_in; (void)out_size; (void)ws_size;
    const float* k_vec      = (const float*)d_in[0];
    const float* k_vec1     = (const float*)d_in[1];
    const float* k_vec2     = (const float*)d_in[2];
    const float* offset_vec = (const float*)d_in[3];
    const float* d_vecs     = (const float*)d_in[4];
    const float* pre_vecs   = (const float*)d_in[5];
    const float* cur_d_vec  = (const float*)d_in[6];
    const float* box_W1 = (const float*)d_in[7];  const float* box_b1 = (const float*)d_in[8];
    const float* box_W2 = (const float*)d_in[9];  const float* box_b2 = (const float*)d_in[10];
    const float* off_W1 = (const float*)d_in[11]; const float* off_b1 = (const float*)d_in[12];
    const float* off_W2 = (const float*)d_in[13]; const float* off_b2 = (const float*)d_in[14];
    const float* ko_W1  = (const float*)d_in[15]; const float* ko_b1  = (const float*)d_in[16];
    const float* ko_W2  = (const float*)d_in[17]; const float* ko_b2  = (const float*)d_in[18];
    const float* Wm1 = (const float*)d_in[19]; const float* bm1 = (const float*)d_in[20];
    const float* Wm2 = (const float*)d_in[21]; const float* bm2 = (const float*)d_in[22];
    const float* Wh1 = (const float*)d_in[23]; const float* bh1 = (const float*)d_in[24];
    const float* Wh2 = (const float*)d_in[25]; const float* bh2 = (const float*)d_in[26];
    const float* Wx1 = (const float*)d_in[27]; const float* bx1 = (const float*)d_in[28];
    const float* Wx2 = (const float*)d_in[29]; const float* bx2 = (const float*)d_in[30];
    const float* up_W1 = (const float*)d_in[31]; const float* up_b1 = (const float*)d_in[32];
    const float* up_W2 = (const float*)d_in[33]; const float* up_b2 = (const float*)d_in[34];

    _Float16* wsH = (_Float16*)d_ws;
    _Float16* BOXW1 = wsH + OFF_BOXW1; _Float16* BOXW2 = wsH + OFF_BOXW2;
    _Float16* OFFW1 = wsH + OFF_OFFW1; _Float16* OFFW2 = wsH + OFF_OFFW2;
    _Float16* KOW1  = wsH + OFF_KOW1;  _Float16* KOW2  = wsH + OFF_KOW2;
    _Float16* WM1f  = wsH + OFF_WM1;   _Float16* WM2f  = wsH + OFF_WM2;
    _Float16* WH1f  = wsH + OFF_WH1;   _Float16* WH2f  = wsH + OFF_WH2;
    _Float16* WX1f  = wsH + OFF_WX1;   _Float16* WX2f  = wsH + OFF_WX2;
    _Float16* UPW1  = wsH + OFF_UPW1;  _Float16* UPW2  = wsH + OFF_UPW2;
    _Float16* OFFBUF = wsH + OFF_OFFBUF;
    _Float16* FEAT   = wsH + OFF_FEAT;
    float* WBUF = (float*)((char*)d_ws + ((OFF_F16END * 2 + 255) & ~(size_t)255));

    auto cvt = [&](const float* src, _Float16* dst, int Kd, int Nc, int Kp, int Np, int nm) {
        int total = nm * Kp * Np;
        int blocks = (total + 255) / 256;
        k_cvt<<<blocks, 256, 0, stream>>>(src, dst, Kd, Nc, Kp, Np, nm);
    };
    cvt(box_W1, BOXW1, 55, 300, 64, 320, 1);
    cvt(box_W2, BOXW2, 300, 100, 320, 112, 1);
    cvt(off_W1, OFFW1, 3, 300, 32, 320, 1);
    cvt(off_W2, OFFW2, 300, 55, 320, 64, 1);
    cvt(ko_W1, KOW1, 165, 300, 192, 320, 1);
    cvt(ko_W2, KOW2, 300, 1, 320, 16, 1);
    cvt(Wm1, WM1f, 200, 300, 224, 320, 6);
    cvt(Wm2, WM2f, 300, 100, 320, 112, 6);
    cvt(Wh1, WH1f, 100, 300, 128, 320, 6);
    cvt(Wh2, WH2f, 300, 100, 320, 112, 6);
    cvt(Wx1, WX1f, 100, 300, 128, 320, 6);
    cvt(Wx2, WX2f, 300, 100, 320, 112, 6);
    cvt(up_W1, UPW1, 700, 300, 704, 320, 1);
    cvt(up_W2, UPW2, 300, 100, 320, 112, 1);

    k_box<<<NN / 32, 64, 0, stream>>>(k_vec, BOXW1, box_b1, BOXW2, box_b2, FEAT);
    k_off<<<NN / 32, 64, 0, stream>>>(offset_vec, OFFW1, off_b1, OFFW2, off_b2, OFFBUF);
    k_ko<<<NN / 32, 64, 0, stream>>>(k_vec1, k_vec2, OFFBUF, KOW1, ko_b1, KOW2, ko_b2, WBUF);
    k_rel<<<dim3(NN / 32, 6), 64, 0, stream>>>(cur_d_vec, d_vecs, pre_vecs, WBUF,
                                               WM1f, bm1, WM2f, bm2,
                                               WH1f, bh1, WH2f, bh2,
                                               WX1f, bx1, WX2f, bx2, FEAT);
    k_up<<<NN / 16, 32, 0, stream>>>(FEAT, UPW1, up_b1, UPW2, up_b2, (float*)d_out);
}
